// SegmentManager_3908420240238
// MI455X (gfx1250) — matrix operand order repaired and checked by reading, not compiled, not run on hardware
//
#include <hip/hip_runtime.h>
#include <hip/hip_bf16.h>

typedef __attribute__((ext_vector_type(16))) _Float16 v16h;
typedef __attribute__((ext_vector_type(8)))  float    v8f;

#define E_    8
#define H_    128
#define DIN_  124
#define KD_   128      // DIN padded
#define DOUT_ 59
#define DO_PAD_ 64
#define EMB_  64
#define SH3_  48
#define ROWS_ 64       // points per block
#define RT_   4        // 16-row subtiles per block

// ---------------------------------------------------------------------------
// Kernel 1: transpose + f32->f16 convert expert weights into d_ws.
// Layouts:  W1T[e][h][k]  (k = padded DIN, 128)   e*H*128
//           W2T[e][h][k]  (k = H, 128)
//           W3T[e][o][k]  (o padded to 64, k = H) e*64*128
// ---------------------------------------------------------------------------
__global__ __launch_bounds__(256) void wprep_kernel(
    const float* __restrict__ W1, const float* __restrict__ W2,
    const float* __restrict__ W3,
    _Float16* __restrict__ W1T, _Float16* __restrict__ W2T,
    _Float16* __restrict__ W3T)
{
  const int S1 = E_ * H_ * KD_;        // 131072
  const int S2 = E_ * H_ * H_;         // 131072
  const int S3 = E_ * DO_PAD_ * H_;    // 65536
  int j = blockIdx.x * blockDim.x + threadIdx.x;
  if (j < S1) {
    int e = j >> 14, rem = j & 16383, h = rem >> 7, d = rem & 127;
    float v = (d < DIN_) ? W1[(e * DIN_ + d) * H_ + h] : 0.0f;
    W1T[j] = (_Float16)v;
  } else if (j < S1 + S2) {
    int k = j - S1;
    int e = k >> 14, rem = k & 16383, h = rem >> 7, d = rem & 127;
    W2T[k] = (_Float16)W2[(e * H_ + d) * H_ + h];
  } else if (j < S1 + S2 + S3) {
    int k = j - S1 - S2;
    int e = k >> 13, rem = k & 8191, o = rem >> 7, d = rem & 127;
    float v = (o < DOUT_) ? W3[(e * H_ + d) * DOUT_ + o] : 0.0f;
    W3T[k] = (_Float16)v;
  }
}

// ---------------------------------------------------------------------------
// Kernel 2: route points into per-expert buckets; also emit `active` output.
// (atomic slot order varies, but per-row results depend only on that row's
//  fixed K-ordered dot products -> output values are deterministic)
// ---------------------------------------------------------------------------
__global__ __launch_bounds__(256) void bucket_kernel(
    const int* __restrict__ seg_ids, const float* __restrict__ timev,
    const float* __restrict__ t_start, const float* __restrict__ t_end,
    int* __restrict__ counts, int* __restrict__ buckets,
    float* __restrict__ active_out, int N)
{
  int g = blockIdx.x * blockDim.x + threadIdx.x;
  if (g >= N) return;
  int e = seg_ids[g];
  int slot = atomicAdd(&counts[e], 1);
  buckets[e * N + slot] = g;
  float t0 = timev[0];
  bool act = (t0 >= t_start[g]) && (t0 < t_end[g]);
  active_out[g] = act ? 1.0f : 0.0f;
}

// ---------------------------------------------------------------------------
// WMMA fragment builders (ISA 7.12.2, wave32, 16-bit A / B, 32-bit C/D)
// ---------------------------------------------------------------------------
__device__ inline v16h load_A_frag(const _Float16* __restrict__ S /*16x128 row-major*/,
                                   int lane, int kb)
{
  v16h a;
  const int M = lane & 15;
  const int khalf = (lane >> 4) << 3;           // 0 or 8
  const _Float16* row = S + M * 128 + kb * 32;
#pragma unroll
  for (int v = 0; v < 8; ++v) {
    int k = ((v >= 4) ? 16 : 0) + khalf + 2 * (v & 3);
    a[2 * v]     = row[k];
    a[2 * v + 1] = row[k + 1];
  }
  return a;
}

__device__ inline v16h load_B_frag(const _Float16* __restrict__ WTcol /* [128] K-major */,
                                   int lane, int kb)
{
  v16h b;
  const int kbase = (lane >> 4) << 3;           // 0 or 8
  const _Float16* p = WTcol + kb * 32 + kbase;
#pragma unroll
  for (int v = 0; v < 8; ++v) {
    int k = ((v >= 4) ? 16 : 0) + 2 * (v & 3);
    b[2 * v]     = p[k];
    b[2 * v + 1] = p[k + 1];
  }
  return b;
}

// ---------------------------------------------------------------------------
// Kernel 3: per-expert MLP.  Block = 256 thr (8 waves), 64 points per block.
// B fragments held in registers across 4 row-subtiles (4x weight reuse).
// LDS: bufA holds X, then H2 (aliased); bufB holds H1.
// ---------------------------------------------------------------------------
__global__ __launch_bounds__(256) void mlp_kernel(
    const float* __restrict__ means, const float* __restrict__ scales,
    const float* __restrict__ rots,  const float* __restrict__ opacity,
    const float* __restrict__ shs,   const float* __restrict__ timev,
    const float* __restrict__ emb,
    const unsigned char* __restrict__ is_static,
    const float* __restrict__ t_start, const float* __restrict__ t_end,
    const float* __restrict__ xyz_s, const float* __restrict__ scl_s,
    const float* __restrict__ rot_s, const float* __restrict__ opa_s,
    const float* __restrict__ shs_s,
    const int* __restrict__ counts, const int* __restrict__ buckets,
    const _Float16* __restrict__ W1T, const _Float16* __restrict__ W2T,
    const _Float16* __restrict__ W3T,
    const float* __restrict__ b1, const float* __restrict__ b2,
    const float* __restrict__ b3,
    float* __restrict__ out, int N, int TILES)
{
  const int e = blockIdx.x / TILES;
  const int t = blockIdx.x % TILES;
  const int cnt = counts[e];
  if (t * ROWS_ >= cnt) return;

  __shared__ int      gidx[ROWS_];
  __shared__ _Float16 bufA[ROWS_ * 128];   // X, later H2
  __shared__ _Float16 bufB[ROWS_ * 128];   // H1
  __shared__ float    Bs [ROWS_ * 60];     // base attrs (f32)

  const int tid  = threadIdx.x;
  const int lane = tid & 31;
  const int wave = tid >> 5;

  if (tid < ROWS_) {
    int r = t * ROWS_ + tid;
    if (r >= cnt) r = cnt - 1;              // duplicate row (masked at store)
    gidx[tid] = buckets[e * N + r];
  }
  __syncthreads();

  // ---- gather MLP input X (f16, 64x128, cols 124..127 zero) ----------------
  for (int i = tid; i < ROWS_ * 128; i += 256) {
    int r = i >> 7, c = i & 127;
    int g = gidx[r];
    float v = 0.0f;
    if      (c < 3)   v = means[g * 3 + c];
    else if (c < 6)   v = scales[g * 3 + (c - 3)];
    else if (c < 10)  v = rots[g * 4 + (c - 6)];
    else if (c == 10) v = opacity[g];
    else if (c == 11) v = timev[g];
    else if (c < 76)  v = emb[g * EMB_ + (c - 12)];
    else if (c < 124) v = shs[g * SH3_ + (c - 76)];
    bufA[i] = (_Float16)v;
  }
  // ---- gather base attributes (f32, 64x59 padded to 60) --------------------
  for (int i = tid; i < ROWS_ * 60; i += 256) {
    int r = i / 60, c = i % 60;
    int g = gidx[r];
    float v = 0.0f;
    if      (c < 3)   v = means[g * 3 + c];
    else if (c < 6)   v = scales[g * 3 + (c - 3)];
    else if (c < 10)  v = rots[g * 4 + (c - 6)];
    else if (c == 10) v = opacity[g];
    else if (c < 59)  v = shs[g * SH3_ + (c - 11)];
    Bs[i] = v;
  }
  __syncthreads();

  const int Mbase = (lane >= 16) ? 8 : 0;

  // ---- layer 1: h1 = relu(X * W1 + b1); wave owns 16 cols, 4 row subtiles --
  {
    const int col = wave * 16 + (lane & 15);
    const float bv = b1[e * H_ + col];
    v8f acc[RT_];
#pragma unroll
    for (int rt = 0; rt < RT_; ++rt)
#pragma unroll
      for (int v = 0; v < 8; ++v) acc[rt][v] = bv;
    const _Float16* WTc = W1T + (size_t)(e * H_ + col) * 128;
#pragma unroll
    for (int kb = 0; kb < 4; ++kb) {
      v16h b = load_B_frag(WTc, lane, kb);
#pragma unroll
      for (int rt = 0; rt < RT_; ++rt) {
        v16h a = load_A_frag(bufA + rt * 16 * 128, lane, kb);
        acc[rt] = __builtin_amdgcn_wmma_f32_16x16x32_f16(false, a, false, b,
                                                         (short)0, acc[rt],
                                                         false, false);
      }
    }
#pragma unroll
    for (int rt = 0; rt < RT_; ++rt)
#pragma unroll
      for (int v = 0; v < 8; ++v) {
        float r = acc[rt][v] > 0.0f ? acc[rt][v] : 0.0f;
        bufB[(rt * 16 + v + Mbase) * 128 + col] = (_Float16)r;
      }
  }
  __syncthreads();

  // ---- layer 2: h2 = relu(h1 * W2 + b2); X no longer needed -> write bufA --
  {
    const int col = wave * 16 + (lane & 15);
    const float bv = b2[e * H_ + col];
    v8f acc[RT_];
#pragma unroll
    for (int rt = 0; rt < RT_; ++rt)
#pragma unroll
      for (int v = 0; v < 8; ++v) acc[rt][v] = bv;
    const _Float16* WTc = W2T + (size_t)(e * H_ + col) * 128;
#pragma unroll
    for (int kb = 0; kb < 4; ++kb) {
      v16h b = load_B_frag(WTc, lane, kb);
#pragma unroll
      for (int rt = 0; rt < RT_; ++rt) {
        v16h a = load_A_frag(bufB + rt * 16 * 128, lane, kb);
        acc[rt] = __builtin_amdgcn_wmma_f32_16x16x32_f16(false, a, false, b,
                                                         (short)0, acc[rt],
                                                         false, false);
      }
    }
    __syncthreads();   // all waves done reading X from bufA
#pragma unroll
    for (int rt = 0; rt < RT_; ++rt)
#pragma unroll
      for (int v = 0; v < 8; ++v) {
        float r = acc[rt][v] > 0.0f ? acc[rt][v] : 0.0f;
        bufA[(rt * 16 + v + Mbase) * 128 + col] = (_Float16)r;
      }
  }
  __syncthreads();

  // ---- layer 3: delta = h2 * W3 + b3; 4 col-tiles x 4 row-subtiles over
  //      8 waves (2 tiles per wave) -------------------------------------------
  {
    const int ct = wave & 3;                    // column tile 0..3
    const int rt0 = (wave >> 2) * 2;            // row subtiles {rt0, rt0+1}
    const int col = ct * 16 + (lane & 15);      // output column o (0..63)
    const float bv = (col < DOUT_) ? b3[e * DOUT_ + col] : 0.0f;
    v8f acc[2];
#pragma unroll
    for (int i = 0; i < 2; ++i)
#pragma unroll
      for (int v = 0; v < 8; ++v) acc[i][v] = bv;
    const _Float16* WTc = W3T + (size_t)(e * DO_PAD_ + col) * 128;
#pragma unroll
    for (int kb = 0; kb < 4; ++kb) {
      v16h b = load_B_frag(WTc, lane, kb);
#pragma unroll
      for (int i = 0; i < 2; ++i) {
        v16h a = load_A_frag(bufA + (rt0 + i) * 16 * 128, lane, kb);
        acc[i] = __builtin_amdgcn_wmma_f32_16x16x32_f16(false, a, false, b,
                                                        (short)0, acc[i],
                                                        false, false);
      }
    }
    if (col < DOUT_) {
      const float t0 = timev[0];
#pragma unroll
      for (int i = 0; i < 2; ++i) {
        const int rt = rt0 + i;
#pragma unroll
        for (int v = 0; v < 8; ++v) {
          const int M = rt * 16 + v + Mbase;    // row within block tile
          const int r = t * ROWS_ + M;
          if (r < cnt) {
            const int g = gidx[M];
            const bool act = (t0 >= t_start[g]) && (t0 < t_end[g]);
            const bool sta = (is_static[g] != 0);
            float val;
            if (act && sta) {
              if      (col < 3)   val = xyz_s[g * 3 + col];
              else if (col < 6)   val = scl_s[g * 3 + (col - 3)];
              else if (col < 10)  val = rot_s[g * 4 + (col - 6)];
              else if (col == 10) val = opa_s[g];
              else                val = shs_s[g * SH3_ + (col - 11)];
            } else if (act) {
              val = Bs[M * 60 + col] + acc[i][v];
            } else {
              val = Bs[M * 60 + col];
            }
            if (col == 10 && !act) val = -100.0f;
            out[(size_t)g * DOUT_ + col] = val;
          }
        }
      }
    }
  }
}

// ---------------------------------------------------------------------------
extern "C" void kernel_launch(void* const* d_in, const int* in_sizes, int n_in,
                              void* d_out, int out_size, void* d_ws, size_t ws_size,
                              hipStream_t stream)
{
  const float* means    = (const float*)d_in[0];
  const float* scales   = (const float*)d_in[1];
  const float* rots     = (const float*)d_in[2];
  const float* opacity  = (const float*)d_in[3];
  const float* shs      = (const float*)d_in[4];
  const float* timev    = (const float*)d_in[5];
  const float* emb      = (const float*)d_in[6];
  const int*   seg_ids  = (const int*)d_in[7];
  const unsigned char* is_static = (const unsigned char*)d_in[8];
  const float* t_start  = (const float*)d_in[9];
  const float* t_end    = (const float*)d_in[10];
  const float* xyz_s    = (const float*)d_in[11];
  const float* scl_s    = (const float*)d_in[12];
  const float* rot_s    = (const float*)d_in[13];
  const float* opa_s    = (const float*)d_in[14];
  const float* shs_s    = (const float*)d_in[15];
  const float* W1       = (const float*)d_in[16];
  const float* b1       = (const float*)d_in[17];
  const float* W2       = (const float*)d_in[18];
  const float* b2       = (const float*)d_in[19];
  const float* W3       = (const float*)d_in[20];
  const float* b3       = (const float*)d_in[21];

  const int N = in_sizes[0] / 3;

  // workspace carve-up (256B aligned sections)
  char* ws = (char*)d_ws;
  int*      counts  = (int*)ws;                                   // 8 ints
  int*      buckets = (int*)(ws + 256);                           // E*N ints
  size_t    boff    = 256 + (size_t)E_ * N * sizeof(int);
  boff = (boff + 255) & ~(size_t)255;
  _Float16* W1T = (_Float16*)(ws + boff);                         // E*H*128
  _Float16* W2T = W1T + (size_t)E_ * H_ * KD_;
  _Float16* W3T = W2T + (size_t)E_ * H_ * H_;

  float* out        = (float*)d_out;                    // [N,59]
  float* active_out = out + (size_t)N * DOUT_;          // [N] (tuple 2nd elem)

  hipMemsetAsync(counts, 0, E_ * sizeof(int), stream);

  const int wtotal = E_ * H_ * KD_ + E_ * H_ * H_ + E_ * DO_PAD_ * H_;
  wprep_kernel<<<(wtotal + 255) / 256, 256, 0, stream>>>(W1, W2, W3, W1T, W2T, W3T);

  bucket_kernel<<<(N + 255) / 256, 256, 0, stream>>>(
      seg_ids, timev, t_start, t_end, counts, buckets, active_out, N);

  const int TILES = N / ROWS_ + 1;
  mlp_kernel<<<E_ * TILES, 256, 0, stream>>>(
      means, scales, rots, opacity, shs, timev, emb, is_static,
      t_start, t_end, xyz_s, scl_s, rot_s, opa_s, shs_s,
      counts, buckets, W1T, W2T, W3T, b1, b2, b3, out, N, TILES);
}
